// HiViT_46084999086679
// MI455X (gfx1250) — matrix operand order repaired and checked by reading, not compiled, not run on hardware
//
#include <hip/hip_runtime.h>
#include <math.h>

// ---------------------------------------------------------------------------
// Types for CDNA5 WMMA (wave32, v_wmma_f32_16x16x32_bf16)
// ---------------------------------------------------------------------------
typedef __bf16 bf16_t;
typedef __attribute__((ext_vector_type(16))) __bf16 v16bf;
typedef __attribute__((ext_vector_type(8)))  __bf16 v8bf;
typedef __attribute__((ext_vector_type(8)))  float  v8f;

#define LN_EPS 1e-5f
#define PADM 224   // attention row padding (multiple of 32), 196 -> 224
#define PADK 224   // attention K padding for attn@v (multiple of 32)

__device__ __forceinline__ float gelu_exact(float v) {
    return 0.5f * v * (1.0f + erff(v * 0.70710678118654752f));
}

__device__ __forceinline__ v8f wmma_bf16(v16bf a, v16bf b, v8f c) {
    // 8 args: (neg_a, A, neg_b, B, c_mod, C, reuse_a, reuse_b)
    return __builtin_amdgcn_wmma_f32_16x16x32_bf16(false, a, false, b, (short)0, c,
                                                   false, false);
}

__device__ __forceinline__ v16bf cat16(v8bf lo, v8bf hi) {
    return __builtin_shufflevector(lo, hi, 0, 1, 2, 3, 4, 5, 6, 7,
                                           8, 9, 10, 11, 12, 13, 14, 15);
}

// ---------------------------------------------------------------------------
// Batched WMMA GEMM:  out = A(MxK, bf16) * Bt(NxK, bf16)^T  (+bias,+gelu,+resid)
// Each wave computes a 32x32 tile (2x2 accumulators -> 4 WMMAs per K-step,
// 8 b128 loads per 4 WMMAs).  One 256-thread block (8 waves) covers 32 x 256.
// A buffers are allocated with Mp rows (Mp = round-up-32(M)); K % 32 == 0.
// ---------------------------------------------------------------------------
__global__ void k_gemm(const bf16_t* __restrict__ A, const bf16_t* __restrict__ Bt,
                       const float* __restrict__ bias, const float* resid,
                       float* outF, bf16_t* outB,
                       int M, int N, int K, int Np,
                       long long sA, long long sB, long long sOF, long long sOB,
                       int gelu)
{
    int z = blockIdx.z;
    A  += (long long)z * sA;
    Bt += (long long)z * sB;
    if (outF) outF += (long long)z * sOF;
    if (outB) outB += (long long)z * sOB;

    int lane = threadIdx.x & 31;
    int wave = threadIdx.x >> 5;
    int l15  = lane & 15;
    int hi   = lane >> 4;                 // lane half: 0 or 1
    int aOff = hi * 8;                    // A: K-interleaved halves (ISA 16-bit A layout)
    int bOff = hi * 8;                    // B: K-interleaved halves (matches A)

    int mBase = blockIdx.y * 32;
    int nBase = blockIdx.x * 256 + wave * 32;

    const bf16_t* aRow0 = A + (size_t)(mBase +      l15) * K;
    const bf16_t* aRow1 = A + (size_t)(mBase + 16 + l15) * K;
    const bf16_t* bRow0;
    const bf16_t* bRow1;
    { int r = nBase + l15;      if (r >= Np) r = 0; bRow0 = Bt + (size_t)r * K; }
    { int r = nBase + 16 + l15; if (r >= Np) r = 0; bRow1 = Bt + (size_t)r * K; }

    v8f acc00 = {}, acc01 = {}, acc10 = {}, acc11 = {};
    for (int k = 0; k < K; k += 32) {
        v16bf a0 = cat16(*(const v8bf*)(aRow0 + k + aOff),
                         *(const v8bf*)(aRow0 + k + 16 + aOff));
        v16bf a1 = cat16(*(const v8bf*)(aRow1 + k + aOff),
                         *(const v8bf*)(aRow1 + k + 16 + aOff));
        v16bf b0 = cat16(*(const v8bf*)(bRow0 + k + bOff),
                         *(const v8bf*)(bRow0 + k + 16 + bOff));
        v16bf b1 = cat16(*(const v8bf*)(bRow1 + k + bOff),
                         *(const v8bf*)(bRow1 + k + 16 + bOff));
        acc00 = wmma_bf16(a0, b0, acc00);
        acc01 = wmma_bf16(a0, b1, acc01);
        acc10 = wmma_bf16(a1, b0, acc10);
        acc11 = wmma_bf16(a1, b1, acc11);
    }

    #pragma unroll
    for (int mt = 0; mt < 2; ++mt) {
        #pragma unroll
        for (int r = 0; r < 8; ++r) {
            int row = mBase + mt * 16 + r + hi * 8;  // C/D layout: VGPR r -> M=r / r+8
            #pragma unroll
            for (int t = 0; t < 2; ++t) {
                int col = nBase + t * 16 + l15;
                float v = (mt == 0) ? ((t == 0) ? acc00[r] : acc01[r])
                                    : ((t == 0) ? acc10[r] : acc11[r]);
                if (col < N) {
                    if (bias) v += bias[col];
                    if (gelu) v = gelu_exact(v);
                }
                if (outF && row < M && col < N) {
                    float o = v;
                    if (resid) o += resid[(size_t)row * N + col];
                    outF[(size_t)row * N + col] = o;
                }
                if (outB && col < Np) {
                    outB[(size_t)row * Np + col] = (col < N) ? (bf16_t)v : (bf16_t)0.f;
                }
            }
        }
    }
}

// ---------------------------------------------------------------------------
// Weight fp32 (KxN) -> transposed bf16 (Np x Kp), zero padded
// ---------------------------------------------------------------------------
__global__ void k_wT(const float* __restrict__ W, bf16_t* __restrict__ out,
                     int K, int N, int Kp, int Np)
{
    int i = blockIdx.x * 256 + threadIdx.x;
    if (i >= Np * Kp) return;
    int n = i / Kp, k = i - n * Kp;
    float v = (n < N && k < K) ? W[(size_t)k * N + n] : 0.f;
    out[i] = (bf16_t)v;
}

// ---------------------------------------------------------------------------
// LayerNorm (fp32 in) fused with bf16 conversion
// ---------------------------------------------------------------------------
__global__ void k_ln_bf16(const float* __restrict__ X, const float* __restrict__ g,
                          const float* __restrict__ b, bf16_t* __restrict__ out,
                          int C, int Cp)
{
    int row = blockIdx.x;
    int tid = threadIdx.x;
    const float* x = X + (size_t)row * C;
    bf16_t* o = out + (size_t)row * Cp;
    __shared__ float s1[256], s2[256];
    float a = 0.f, q = 0.f;
    for (int c = tid; c < C; c += 256) { float v = x[c]; a += v; q += v * v; }
    s1[tid] = a; s2[tid] = q; __syncthreads();
    for (int w = 128; w > 0; w >>= 1) {
        if (tid < w) { s1[tid] += s1[tid + w]; s2[tid] += s2[tid + w]; }
        __syncthreads();
    }
    float mean = s1[0] / C;
    float var  = s2[0] / C - mean * mean;
    float inv  = rsqrtf(var + LN_EPS);
    for (int c = tid; c < Cp; c += 256) {
        float v = (c < C) ? (x[c] - mean) * inv * g[c] + b[c] : 0.f;
        o[c] = (bf16_t)v;
    }
}

// ---------------------------------------------------------------------------
// Patch-merge gather (2x2 -> 1, channels x4) fused with LayerNorm -> bf16
// ---------------------------------------------------------------------------
__global__ void k_merge_ln(const float* __restrict__ Y, const float* __restrict__ g,
                           const float* __restrict__ b, bf16_t* __restrict__ out,
                           int dim, int Sin)
{
    int Sout = Sin >> 1;
    int ro   = blockIdx.x;
    int io   = ro % (Sout * Sout);
    int tok  = ro / (Sout * Sout);
    int ihO  = io / Sout, iwO = io % Sout;
    int C    = 4 * dim;
    int tid  = threadIdx.x;
    const float* src[4];
    #pragma unroll
    for (int p = 0; p < 4; ++p) {
        int ih = 2 * ihO + (p & 1);
        int iw = 2 * iwO + ((p >> 1) & 1);
        src[p] = Y + ((size_t)tok * (Sin * Sin) + ih * Sin + iw) * dim;
    }
    __shared__ float s1[256], s2[256];
    float a = 0.f, q = 0.f;
    for (int c = tid; c < C; c += 256) {
        float v = src[c / dim][c % dim];
        a += v; q += v * v;
    }
    s1[tid] = a; s2[tid] = q; __syncthreads();
    for (int w = 128; w > 0; w >>= 1) {
        if (tid < w) { s1[tid] += s1[tid + w]; s2[tid] += s2[tid + w]; }
        __syncthreads();
    }
    float mean = s1[0] / C;
    float var  = s2[0] / C - mean * mean;
    float inv  = rsqrtf(var + LN_EPS);
    bf16_t* o = out + (size_t)ro * C;
    for (int c = tid; c < C; c += 256) {
        float v = src[c / dim][c % dim];
        o[c] = (bf16_t)((v - mean) * inv * g[c] + b[c]);
    }
}

// ---------------------------------------------------------------------------
// Patch embed: 4x4/4 conv (3->128) + bias, rearranged to (B,196,4,4,128), + LN
// ---------------------------------------------------------------------------
__global__ void k_patch_embed(const float* __restrict__ x, const float* __restrict__ w,
                              const float* __restrict__ pb, const float* __restrict__ ng,
                              const float* __restrict__ nb, float* __restrict__ y)
{
    int idx = blockIdx.x;                       // B*56*56
    int ow  = idx % 56;
    int oh  = (idx / 56) % 56;
    int b   = idx / (56 * 56);
    __shared__ float patch[48];
    __shared__ float s1[128], s2[128];
    int t = threadIdx.x;                        // 128
    if (t < 48) {
        int kw = t & 3, kh = (t >> 2) & 3, ci = t >> 4;
        patch[t] = x[(((size_t)b * 3 + ci) * 224 + (oh * 4 + kh)) * 224 + (ow * 4 + kw)];
    }
    __syncthreads();
    float acc = pb[t];
    const float* wr = w + (size_t)t * 48;
    #pragma unroll
    for (int i = 0; i < 48; ++i) acc += patch[i] * wr[i];
    s1[t] = acc; s2[t] = acc * acc; __syncthreads();
    for (int wd = 64; wd > 0; wd >>= 1) {
        if (t < wd) { s1[t] += s1[t + wd]; s2[t] += s2[t + wd]; }
        __syncthreads();
    }
    float mean = s1[0] / 128.f;
    float var  = s2[0] / 128.f - mean * mean;
    float inv  = rsqrtf(var + LN_EPS);
    float o    = (acc - mean) * inv * ng[t] + nb[t];
    int hp = oh >> 2, ihh = oh & 3, wp = ow >> 2, iww = ow & 3;
    size_t r = ((size_t)(b * 196 + hp * 14 + wp)) * 16 + ihh * 4 + iww;
    y[r * 128 + t] = o;
}

// ---------------------------------------------------------------------------
// qkv (M3 x 1536 fp32) -> q,k bf16 (BH,PADM,64) [q scaled 1/8], v^T (BH,64,PADK)
// Pad rows/cols zero-filled so WMMA K-reductions stay exact.
// ---------------------------------------------------------------------------
__global__ void k_qkv_split(const float* __restrict__ qkv,
                            bf16_t* __restrict__ q, bf16_t* __restrict__ k,
                            bf16_t* __restrict__ vt)
{
    int bn = blockIdx.x;                        // B*8*PADM  (PADM == PADK == 224)
    int n  = bn % PADM;
    int bh = bn / PADM;
    int h  = bh & 7;
    int b  = bh >> 3;
    int d  = threadIdx.x;                       // 64
    float qv = 0.f, kv = 0.f, vv = 0.f;
    if (n < 196) {
        size_t row = ((size_t)(b * 196 + n)) * 1536;
        qv = qkv[row +        h * 64 + d] * 0.125f;   // hd^-0.5, hd=64
        kv = qkv[row +  512 + h * 64 + d];
        vv = qkv[row + 1024 + h * 64 + d];
    }
    q[((size_t)bh * PADM + n) * 64 + d] = (bf16_t)qv;
    k[((size_t)bh * PADM + n) * 64 + d] = (bf16_t)kv;
    vt[((size_t)bh * 64 + d) * PADK + n] = (bf16_t)vv;
}

// ---------------------------------------------------------------------------
// Softmax with relative-position bias; writes bf16 attn padded to PADM x PADK
// rp_index(n,m) = (i1-i2+13)*27 + (j1-j2+13), bias = rpb[idx, h]
// ---------------------------------------------------------------------------
__global__ void k_softmax(const float* __restrict__ S, const float* __restrict__ rpb,
                          bf16_t* __restrict__ out)
{
    int bn = blockIdx.x;                        // B*8*PADM
    int n  = bn % PADM;
    int bh = bn / PADM;
    int h  = bh & 7;
    bf16_t* o = out + ((size_t)bh * PADM + n) * PADK;
    int tid = threadIdx.x;                      // 256
    if (n >= 196) {
        for (int m = tid; m < PADK; m += 256) o[m] = (bf16_t)0.f;
        return;
    }
    const float* srow = S + ((size_t)bh * 196 + n) * 196;
    int i1 = n / 14, j1 = n % 14;
    __shared__ float sh[256];
    float myv = 0.f;
    float v   = -1e30f;
    if (tid < 196) {
        int i2  = tid / 14, j2 = tid % 14;
        int idx = (i1 - i2 + 13) * 27 + (j1 - j2 + 13);
        myv = srow[tid] + rpb[idx * 8 + h];
        v   = myv;
    }
    sh[tid] = v; __syncthreads();
    for (int w = 128; w > 0; w >>= 1) {
        if (tid < w) sh[tid] = fmaxf(sh[tid], sh[tid + w]);
        __syncthreads();
    }
    float mx = sh[0]; __syncthreads();
    float e = (tid < 196) ? __expf(myv - mx) : 0.f;
    sh[tid] = e; __syncthreads();
    for (int w = 128; w > 0; w >>= 1) {
        if (tid < w) sh[tid] += sh[tid + w];
        __syncthreads();
    }
    float inv = 1.f / sh[0];
    if (tid < PADK) o[tid] = (tid < 196) ? (bf16_t)(e * inv) : (bf16_t)0.f;
}

// ---------------------------------------------------------------------------
// Combine per-head outputs (BH,PADM,64) -> (M3, 512) bf16 with col = h*64+d
// ---------------------------------------------------------------------------
__global__ void k_combine(const bf16_t* __restrict__ hout, bf16_t* __restrict__ aout)
{
    int r = blockIdx.x;                         // 12544
    int c = threadIdx.x;                        // 512
    int b = r / 196, n = r % 196;
    int h = c >> 6, d = c & 63;
    aout[(size_t)r * 512 + c] = hout[((size_t)(b * 8 + h) * PADM + n) * 64 + d];
}

__global__ void k_add_pos(float* y, const float* __restrict__ pos)
{
    size_t i = (size_t)blockIdx.x * 256 + threadIdx.x;
    if (i < (size_t)12544 * 512) y[i] += pos[i % (196 * 512)];
}

__global__ void k_mean(const float* __restrict__ y, float* __restrict__ out)
{
    int b = blockIdx.x;                         // 64
    int c = threadIdx.x;                        // 512
    float s = 0.f;
    for (int t = 0; t < 196; ++t) s += y[((size_t)(b * 196 + t)) * 512 + c];
    out[b * 512 + c] = s * (1.f / 196.f);
}

// ---------------------------------------------------------------------------
// Host side
// ---------------------------------------------------------------------------
struct HParams {
    const float* x;
    const float *pe_w, *pe_b, *pe_ng, *pe_nb;
    const float *s_n2g[16], *s_n2b[16], *s_f1w[16], *s_f1b[16], *s_f2w[16], *s_f2b[16];
    const float *m_ng[2], *m_nb[2], *m_rw[2];
    const float *n1g[19], *n1b[19], *qkvw[19], *qkvb[19], *projw[19], *projb[19], *rpb[19];
    const float *n2g[19], *n2b[19], *f1w[19], *f1b[19], *f2w[19], *f2b[19];
    const float *pos, *fcg, *fcb, *headw, *headb;
};

static void fill_insertion(HParams& P, void* const* d)
{
    int i = 0;
    auto F = [&]() { return (const float*)d[i++]; };
    P.x = F();
    P.pe_w = F(); P.pe_b = F(); P.pe_ng = F(); P.pe_nb = F();
    for (int st = 0; st < 2; ++st) {
        for (int s = 0; s < 8; ++s) {
            int k = st * 8 + s;
            P.s_n2g[k] = F(); P.s_n2b[k] = F();
            P.s_f1w[k] = F(); P.s_f1b[k] = F();
            P.s_f2w[k] = F(); P.s_f2b[k] = F();
        }
        P.m_ng[st] = F(); P.m_nb[st] = F(); P.m_rw[st] = F();
    }
    for (int l = 0; l < 19; ++l) {
        P.n1g[l] = F(); P.n1b[l] = F();
        P.qkvw[l] = F(); P.qkvb[l] = F();
        P.projw[l] = F(); P.projb[l] = F();
        P.rpb[l] = F();
        P.n2g[l] = F(); P.n2b[l] = F();
        P.f1w[l] = F(); P.f1b[l] = F();
        P.f2w[l] = F(); P.f2b[l] = F();
    }
    P.pos = F(); P.fcg = F(); P.fcb = F(); P.headw = F(); P.headb = F();
}

static void fill_sorted(HParams& P, void* const* d)
{
    int i = 0;
    auto F = [&]() { return (const float*)d[i++]; };
    P.fcb = F(); P.fcg = F(); P.headb = F(); P.headw = F();
    for (int l = 0; l < 19; ++l) {
        P.f1b[l] = F(); P.f1w[l] = F(); P.f2b[l] = F(); P.f2w[l] = F();
        P.n1b[l] = F(); P.n1g[l] = F(); P.n2b[l] = F(); P.n2g[l] = F();
        P.projb[l] = F(); P.projw[l] = F();
        P.qkvb[l] = F(); P.qkvw[l] = F();
        P.rpb[l] = F();
    }
    P.pe_b = F(); P.pe_nb = F(); P.pe_ng = F(); P.pe_w = F();
    P.pos = F();
    for (int st = 0; st < 2; ++st) {
        for (int s = 0; s < 8; ++s) {
            int k = st * 8 + s;
            P.s_f1b[k] = F(); P.s_f1w[k] = F(); P.s_f2b[k] = F(); P.s_f2w[k] = F();
            P.s_n2b[k] = F(); P.s_n2g[k] = F();
        }
        P.m_nb[st] = F(); P.m_ng[st] = F(); P.m_rw[st] = F();
    }
    P.x = F();
}

static void gemm(hipStream_t s, const bf16_t* A, const bf16_t* Bt,
                 const float* bias, const float* resid, float* outF, bf16_t* outB,
                 int M, int N, int K, int batch,
                 long long sA, long long sB, long long sOF, long long sOB, int gelu)
{
    int Mp = (M + 31) & ~31;   // A buffers are allocated with Mp rows
    int Np = (N + 15) & ~15;
    dim3 grid((Np + 255) / 256, Mp / 32, batch);
    k_gemm<<<grid, 256, 0, s>>>(A, Bt, bias, resid, outF, outB,
                                M, N, K, Np, sA, sB, sOF, sOB, gelu);
}

static void wT(hipStream_t s, const float* W, bf16_t* out, int K, int N)
{
    int Kp = (K + 31) & ~31;
    int Np = (N + 15) & ~15;
    int total = Np * Kp;
    k_wT<<<dim3((total + 255) / 256), 256, 0, s>>>(W, out, K, N, Kp, Np);
}

static void ln(hipStream_t s, const float* X, const float* g, const float* b,
               bf16_t* out, int M, int C)
{
    k_ln_bf16<<<dim3(M), 256, 0, s>>>(X, g, b, out, C, C);
}

extern "C" void kernel_launch(void* const* d_in, const int* in_sizes, int n_in,
                              void* d_out, int out_size, void* d_ws, size_t ws_size,
                              hipStream_t stream)
{
    HParams P;
    if (n_in > 0 && in_sizes[0] == 64 * 3 * 224 * 224) fill_insertion(P, d_in);
    else                                               fill_sorted(P, d_in);

    char* base = (char*)d_ws;
    size_t off = 0;
    auto alloc = [&](size_t bytes) -> char* {
        char* p = base + off;
        off = (off + bytes + 255) & ~(size_t)255;
        return p;
    };

    const int Bn = 64;
    const int M1 = Bn * 196 * 16;   // 200704 (stage-1 rows)
    const int M2 = Bn * 196 * 4;    // 50176  (stage-2 rows)
    const int M3 = Bn * 196;        // 12544  (main rows)

    // persistent residual streams (fp32)
    float* yS1 = (float*)alloc((size_t)M1 * 128 * 4);
    float* yS2 = (float*)alloc((size_t)M2 * 256 * 4);
    float* yM  = (float*)alloc((size_t)M3 * 512 * 4);
    size_t poolOff = off;

    // ---- patch embed + LN ----
    k_patch_embed<<<dim3(Bn * 56 * 56), 128, 0, stream>>>(
        P.x, P.pe_w, P.pe_b, P.pe_ng, P.pe_nb, yS1);

    // ---- stem stage 1 (dim 128, hidden 384) ----
    {
        off = poolOff;
        bf16_t* f1T = (bf16_t*)alloc((size_t)384 * 128 * 2);
        bf16_t* f2T = (bf16_t*)alloc((size_t)128 * 384 * 2);
        bf16_t* xln = (bf16_t*)alloc((size_t)M1 * 128 * 2);
        bf16_t* hid = (bf16_t*)alloc((size_t)M1 * 384 * 2);
        for (int s = 0; s < 8; ++s) {
            wT(stream, P.s_f1w[s], f1T, 128, 384);
            wT(stream, P.s_f2w[s], f2T, 384, 128);
            ln(stream, yS1, P.s_n2g[s], P.s_n2b[s], xln, M1, 128);
            gemm(stream, xln, f1T, P.s_f1b[s], nullptr, nullptr, hid,
                 M1, 384, 128, 1, 0, 0, 0, 0, 1);
            gemm(stream, hid, f2T, P.s_f2b[s], yS1, yS1, nullptr,
                 M1, 128, 384, 1, 0, 0, 0, 0, 0);
        }
    }
    // ---- merge 1: (4x4,128) -> (2x2,256) ----
    {
        off = poolOff;
        bf16_t* rwT = (bf16_t*)alloc((size_t)256 * 512 * 2);
        bf16_t* mrg = (bf16_t*)alloc((size_t)M2 * 512 * 2);
        wT(stream, P.m_rw[0], rwT, 512, 256);
        k_merge_ln<<<dim3(M2), 256, 0, stream>>>(yS1, P.m_ng[0], P.m_nb[0], mrg, 128, 4);
        gemm(stream, mrg, rwT, nullptr, nullptr, yS2, nullptr,
             M2, 256, 512, 1, 0, 0, 0, 0, 0);
    }
    // ---- stem stage 2 (dim 256, hidden 768) ----
    {
        off = poolOff;
        bf16_t* f1T = (bf16_t*)alloc((size_t)768 * 256 * 2);
        bf16_t* f2T = (bf16_t*)alloc((size_t)256 * 768 * 2);
        bf16_t* xln = (bf16_t*)alloc((size_t)M2 * 256 * 2);
        bf16_t* hid = (bf16_t*)alloc((size_t)M2 * 768 * 2);
        for (int s = 8; s < 16; ++s) {
            wT(stream, P.s_f1w[s], f1T, 256, 768);
            wT(stream, P.s_f2w[s], f2T, 768, 256);
            ln(stream, yS2, P.s_n2g[s], P.s_n2b[s], xln, M2, 256);
            gemm(stream, xln, f1T, P.s_f1b[s], nullptr, nullptr, hid,
                 M2, 768, 256, 1, 0, 0, 0, 0, 1);
            gemm(stream, hid, f2T, P.s_f2b[s], yS2, yS2, nullptr,
                 M2, 256, 768, 1, 0, 0, 0, 0, 0);
        }
    }
    // ---- merge 2: (2x2,256) -> (1x1,512) ----
    {
        off = poolOff;
        bf16_t* rwT = (bf16_t*)alloc((size_t)512 * 1024 * 2);
        bf16_t* mrg = (bf16_t*)alloc((size_t)M3 * 1024 * 2);
        wT(stream, P.m_rw[1], rwT, 1024, 512);
        k_merge_ln<<<dim3(M3), 256, 0, stream>>>(yS2, P.m_ng[1], P.m_nb[1], mrg, 256, 2);
        gemm(stream, mrg, rwT, nullptr, nullptr, yM, nullptr,
             M3, 512, 1024, 1, 0, 0, 0, 0, 0);
    }
    // ---- + pos_embed ----
    k_add_pos<<<dim3((M3 * 512 + 255) / 256), 256, 0, stream>>>(yM, P.pos);

    // ---- 19 main transformer blocks ----
    {
        off = poolOff;
        bf16_t* qkvT = (bf16_t*)alloc((size_t)1536 * 512 * 2);
        bf16_t* projT= (bf16_t*)alloc((size_t)512 * 512 * 2);
        bf16_t* f1T  = (bf16_t*)alloc((size_t)2048 * 512 * 2);
        bf16_t* f2T  = (bf16_t*)alloc((size_t)512 * 2048 * 2);
        bf16_t* xln  = (bf16_t*)alloc((size_t)M3 * 512 * 2);
        float*  qkvF = (float*) alloc((size_t)M3 * 1536 * 4);
        bf16_t* qB   = (bf16_t*)alloc((size_t)512 * PADM * 64 * 2);
        bf16_t* kB   = (bf16_t*)alloc((size_t)512 * PADM * 64 * 2);
        bf16_t* vtB  = (bf16_t*)alloc((size_t)512 * 64 * PADK * 2);
        float*  scF  = (float*) alloc((size_t)512 * 196 * 196 * 4);
        bf16_t* atB  = (bf16_t*)alloc((size_t)512 * PADM * PADK * 2);
        bf16_t* hoB  = (bf16_t*)alloc((size_t)512 * PADM * 64 * 2);
        bf16_t* aoB  = (bf16_t*)alloc((size_t)M3 * 512 * 2);
        bf16_t* mhB  = (bf16_t*)alloc((size_t)M3 * 2048 * 2);

        for (int l = 0; l < 19; ++l) {
            wT(stream, P.qkvw[l], qkvT, 512, 1536);
            wT(stream, P.projw[l], projT, 512, 512);
            wT(stream, P.f1w[l],  f1T,  512, 2048);
            wT(stream, P.f2w[l],  f2T, 2048, 512);

            // attention
            ln(stream, yM, P.n1g[l], P.n1b[l], xln, M3, 512);
            gemm(stream, xln, qkvT, P.qkvb[l], nullptr, qkvF, nullptr,
                 M3, 1536, 512, 1, 0, 0, 0, 0, 0);
            k_qkv_split<<<dim3(512 * PADM), 64, 0, stream>>>(qkvF, qB, kB, vtB);
            // scores = q @ k^T  (batched over B*NH = 512)
            gemm(stream, qB, kB, nullptr, nullptr, scF, nullptr,
                 196, 196, 64, 512,
                 (long long)PADM * 64, (long long)PADM * 64,
                 (long long)196 * 196, 0, 0);
            k_softmax<<<dim3(512 * PADM), 256, 0, stream>>>(scF, P.rpb[l], atB);
            // out = attn @ v
            gemm(stream, atB, vtB, nullptr, nullptr, nullptr, hoB,
                 196, 64, PADK, 512,
                 (long long)PADM * PADK, (long long)64 * PADK,
                 0, (long long)PADM * 64, 0);
            k_combine<<<dim3(M3), 512, 0, stream>>>(hoB, aoB);
            gemm(stream, aoB, projT, P.projb[l], yM, yM, nullptr,
                 M3, 512, 512, 1, 0, 0, 0, 0, 0);

            // mlp
            ln(stream, yM, P.n2g[l], P.n2b[l], xln, M3, 512);
            gemm(stream, xln, f1T, P.f1b[l], nullptr, nullptr, mhB,
                 M3, 2048, 512, 1, 0, 0, 0, 0, 1);
            gemm(stream, mhB, f2T, P.f2b[l], yM, yM, nullptr,
                 M3, 512, 2048, 1, 0, 0, 0, 0, 0);
        }
    }
    // ---- head: mean over tokens, LN, 512 -> 1000 ----
    {
        off = poolOff;
        float*  mout = (float*) alloc((size_t)64 * 512 * 4);
        bf16_t* lnh  = (bf16_t*)alloc((size_t)64 * 512 * 2);
        bf16_t* hT   = (bf16_t*)alloc((size_t)1008 * 512 * 2);
        k_mean<<<dim3(64), 512, 0, stream>>>(yM, mout);
        ln(stream, mout, P.fcg, P.fcb, lnh, 64, 512);
        wT(stream, P.headw, hT, 512, 1000);
        gemm(stream, lnh, hT, P.headb, nullptr, (float*)d_out, nullptr,
             64, 1000, 512, 1, 0, 0, 0, 0, 0);
    }
    (void)ws_size; (void)out_size;
}
